// MambaFull_47820165873728
// MI455X (gfx1250) — matrix operand order repaired and checked by reading, not compiled, not run on hardware
//
#include <hip/hip_runtime.h>
#include <hip/hip_bf16.h>

// ---------------- problem constants ----------------
#define BB    64
#define LL    200
#define DM    1024
#define NLAY  6
#define CITY  200
#define DI    2048      // 2*DM
#define NST   16
#define DTR   64        // DM/16
#define DCV   4
#define MTOK  (BB*LL)   // 12800, multiple of 256
#define EPSV  1e-5f

// ---------------- types ----------------
typedef __attribute__((ext_vector_type(16))) __bf16 bf16x16;
typedef __attribute__((ext_vector_type(8)))  __bf16 bf16x8;
typedef __attribute__((ext_vector_type(8)))  float  f32x8;

union BFFrag { bf16x16 v; bf16x8 h[2]; };

__device__ __forceinline__ unsigned short f32_to_bf16(float f) {
    unsigned u = __float_as_uint(f);
    unsigned r = u + 0x7fffu + ((u >> 16) & 1u);   // round-to-nearest-even
    return (unsigned short)(r >> 16);
}
__device__ __forceinline__ float silu(float x) {
    return x / (1.f + __expf(-x));
}

// ---------------- weight conversion ----------------
__global__ void cvt_bf16_kernel(const float* __restrict__ src,
                                unsigned short* __restrict__ dst, int n) {
    int i = blockIdx.x * blockDim.x + threadIdx.x;
    if (i < n) dst[i] = f32_to_bf16(src[i]);
}

// pad rows per layer: src (layers, rows, cols) -> dst (layers, rowsPad, cols), zero-fill
__global__ void cvt_pad_kernel(const float* __restrict__ src,
                               unsigned short* __restrict__ dst,
                               int rows, int rowsPad, int cols, int total) {
    int i = blockIdx.x * blockDim.x + threadIdx.x;
    if (i >= total) return;
    int c  = i % cols;
    int rl = i / cols;
    int r  = rl % rowsPad;
    int l  = rl / rowsPad;
    float v = (r < rows) ? src[((size_t)l * rows + r) * cols + c] : 0.f;
    dst[i] = f32_to_bf16(v);
}

// ---------------- embedding ----------------
__global__ void embed_kernel(const float* __restrict__ x,
                             const float* __restrict__ w,   // (DM,2)
                             const float* __restrict__ b,
                             float* __restrict__ h, int total) {
    int i = blockIdx.x * blockDim.x + threadIdx.x;
    if (i >= total) return;
    int d = i & (DM - 1);
    int m = i >> 10;
    h[i] = x[m * 2 + 0] * w[d * 2 + 0] + x[m * 2 + 1] * w[d * 2 + 1] + b[d];
}

// ---------------- residual + layernorm -> bf16 ----------------
__global__ __launch_bounds__(256)
void residual_ln_kernel(const float* __restrict__ h, float* __restrict__ res,
                        const float* __restrict__ w, const float* __restrict__ bias,
                        unsigned short* __restrict__ outbf, int first) {
    __shared__ float s1[256], s2[256];
    int m = blockIdx.x;
    const float* hr = h + (size_t)m * DM;
    float* rr = res + (size_t)m * DM;
    float v[4]; float sum = 0.f, sq = 0.f;
#pragma unroll
    for (int j = 0; j < 4; ++j) {
        int idx = threadIdx.x + j * 256;
        float t = hr[idx];
        if (!first) t += rr[idx];
        rr[idx] = t;
        v[j] = t; sum += t; sq += t * t;
    }
    s1[threadIdx.x] = sum; s2[threadIdx.x] = sq; __syncthreads();
    for (int off = 128; off > 0; off >>= 1) {
        if ((int)threadIdx.x < off) {
            s1[threadIdx.x] += s1[threadIdx.x + off];
            s2[threadIdx.x] += s2[threadIdx.x + off];
        }
        __syncthreads();
    }
    float mean = s1[0] * (1.f / DM);
    float var  = s2[0] * (1.f / DM) - mean * mean;
    float rstd = rsqrtf(var + EPSV);
#pragma unroll
    for (int j = 0; j < 4; ++j) {
        int idx = threadIdx.x + j * 256;
        outbf[(size_t)m * DM + idx] = f32_to_bf16((v[j] - mean) * rstd * w[idx] + bias[idx]);
    }
}

// ---------------- final residual + rmsnorm -> bf16 ----------------
__global__ __launch_bounds__(256)
void final_rms_kernel(const float* __restrict__ h, const float* __restrict__ res,
                      const float* __restrict__ w, const float* __restrict__ bias,
                      unsigned short* __restrict__ outbf) {
    __shared__ float s2[256];
    int m = blockIdx.x;
    const float* hr = h + (size_t)m * DM;
    const float* rr = res + (size_t)m * DM;
    float v[4]; float sq = 0.f;
#pragma unroll
    for (int j = 0; j < 4; ++j) {
        int idx = threadIdx.x + j * 256;
        float t = hr[idx] + rr[idx];
        v[j] = t; sq += t * t;
    }
    s2[threadIdx.x] = sq; __syncthreads();
    for (int off = 128; off > 0; off >>= 1) {
        if ((int)threadIdx.x < off) s2[threadIdx.x] += s2[threadIdx.x + off];
        __syncthreads();
    }
    float rms = rsqrtf(s2[0] * (1.f / DM) + EPSV);
#pragma unroll
    for (int j = 0; j < 4; ++j) {
        int idx = threadIdx.x + j * 256;
        outbf[(size_t)m * DM + idx] = f32_to_bf16(v[j] * rms * w[idx] + bias[idx]);
    }
}

// ---------------- WMMA bf16 GEMM: C[M,N] = A[M,K] * W[N,K]^T ----------------
// wave tile 64x64 (4x4 wmma frags, 16 f32x8 accumulators -> 32 FLOP/byte reuse),
// block = 8 waves (4 along M, 2 along N) -> 256(M) x 128(N) block tile.
// M must be a multiple of 256 (12800 here); N padded via weight padding, stores guarded.
__global__ __launch_bounds__(256)
void gemm_bf16_wmma(const __bf16* __restrict__ A, int lda,
                    const __bf16* __restrict__ W, int ldw,
                    float* __restrict__ C, int ldc,
                    int K, int Nreal) {
    int lane = threadIdx.x & 31;
    int wid  = threadIdx.x >> 5;
    int wm = wid >> 1, wn = wid & 1;
    int m0 = blockIdx.y * 256 + wm * 64;
    int n0 = blockIdx.x * 128 + wn * 64;

    int l16  = lane & 15;
    int agrp = (lane >> 4) * 8;    // A: K sub-offset (0 or 8; +16 for upper half)
    int bsel = (lane >> 4) * 16;   // unused; B loads use agrp (same k order as A)

    const __bf16* Ap[4];
    const __bf16* Wp[4];
#pragma unroll
    for (int i = 0; i < 4; ++i) {
        Ap[i] = A + (size_t)(m0 + 16 * i + l16) * lda;
        Wp[i] = W + (size_t)(n0 + 16 * i + l16) * ldw;
    }

    f32x8 acc[4][4];
    const f32x8 zero = {0.f, 0.f, 0.f, 0.f, 0.f, 0.f, 0.f, 0.f};
#pragma unroll
    for (int i = 0; i < 4; ++i)
#pragma unroll
        for (int j = 0; j < 4; ++j) acc[i][j] = zero;

    for (int k = 0; k < K; k += 32) {
        BFFrag a[4], b[4];
#pragma unroll
        for (int i = 0; i < 4; ++i) {
            a[i].h[0] = *(const bf16x8*)(Ap[i] + k + agrp);
            a[i].h[1] = *(const bf16x8*)(Ap[i] + k + 16 + agrp);
        }
#pragma unroll
        for (int j = 0; j < 4; ++j) {
            b[j].h[0] = *(const bf16x8*)(Wp[j] + k + agrp);
            b[j].h[1] = *(const bf16x8*)(Wp[j] + k + 16 + agrp);
        }
#pragma unroll
        for (int i = 0; i < 4; ++i)
#pragma unroll
            for (int j = 0; j < 4; ++j)
                acc[i][j] = __builtin_amdgcn_wmma_f32_16x16x32_bf16(
                    false, a[i].v, false, b[j].v, (short)0, acc[i][j], false, false);
    }

    int col = l16;
    int rhi = (lane >> 4) * 8;
#pragma unroll
    for (int i = 0; i < 4; ++i) {
#pragma unroll
        for (int j = 0; j < 4; ++j) {
            int n = n0 + 16 * j + col;
            if (n < Nreal) {
#pragma unroll
                for (int r = 0; r < 8; ++r) {
                    C[(size_t)(m0 + 16 * i + rhi + r) * ldc + n] = acc[i][j][r];
                }
            }
        }
    }
}

// ---------------- causal depthwise conv (DC=4) + SiLU ----------------
__global__ void conv_silu_kernel(const float* __restrict__ xz,   // (M, 2*DI), xm = [:DI]
                                 const float* __restrict__ cw,   // (DI, DC)
                                 const float* __restrict__ cb,   // (DI)
                                 float* __restrict__ xc,
                                 unsigned short* __restrict__ xcbf, int total) {
    int i = blockIdx.x * blockDim.x + threadIdx.x;
    if (i >= total) return;
    int d = i & (DI - 1);
    int m = i >> 11;
    int t = m % LL, b = m / LL;
    float acc = cb[d];
#pragma unroll
    for (int k = 0; k < DCV; ++k) {
        int tt = t + k - (DCV - 1);
        if (tt >= 0) acc += cw[d * DCV + k] * xz[((size_t)(b * LL + tt)) * (2 * DI) + d];
    }
    float s = silu(acc);
    xc[i] = s;
    xcbf[i] = f32_to_bf16(s);
}

// ---------------- dt_raw slice -> bf16 ----------------
__global__ void dtraw_cvt_kernel(const float* __restrict__ dbl,  // (M,96)
                                 unsigned short* __restrict__ out, int total) {
    int i = blockIdx.x * blockDim.x + threadIdx.x;
    if (i >= total) return;
    int j = i & (DTR - 1);
    int m = i >> 6;
    out[i] = f32_to_bf16(dbl[(size_t)m * 96 + j]);
}

// ---------------- dt bias + softplus (in place) ----------------
__global__ void dt_softplus_kernel(float* __restrict__ dt,
                                   const float* __restrict__ dtb, int total) {
    int i = blockIdx.x * blockDim.x + threadIdx.x;
    if (i >= total) return;
    int d = i & (DI - 1);
    float x = dt[i] + dtb[d];
    dt[i] = (x > 20.f) ? x : log1pf(__expf(x));
}

// ---------------- selective scan + gating epilogue ----------------
__global__ __launch_bounds__(256)
void scan_kernel(const float* __restrict__ dbl,   // (M,96): B at +64, C at +80
                 const float* __restrict__ dt,    // (M,DI)
                 const float* __restrict__ xc,    // (M,DI)
                 const float* __restrict__ xz,    // (M,2*DI), z at +DI
                 const float* __restrict__ A_log, // (DI,NST)
                 const float* __restrict__ Dp,    // (DI)
                 unsigned short* __restrict__ ybf) {
    __shared__ float sB[NST], sC[NST];
    int b = blockIdx.y;
    int d = blockIdx.x * 256 + threadIdx.x;
    float A[NST], s[NST];
#pragma unroll
    for (int n = 0; n < NST; ++n) {
        A[n] = -__expf(A_log[(size_t)d * NST + n]);
        s[n] = 0.f;
    }
    float Dv = Dp[d];
    for (int t = 0; t < LL; ++t) {
        size_t m = (size_t)b * LL + t;
        if (threadIdx.x < 32) {
            float v = dbl[m * 96 + 64 + threadIdx.x];
            if (threadIdx.x < 16) sB[threadIdx.x] = v; else sC[threadIdx.x - 16] = v;
        }
        __syncthreads();
        float dtv = dt[m * DI + d];
        float u   = xc[m * DI + d];
        float du  = dtv * u;
        float y = 0.f;
#pragma unroll
        for (int n = 0; n < NST; ++n) {
            s[n] = s[n] * __expf(dtv * A[n]) + du * sB[n];
            y += s[n] * sC[n];
        }
        float z = xz[m * (2 * DI) + DI + d];
        float out = (y + u * Dv) * silu(z);
        ybf[m * DI + d] = f32_to_bf16(out);
        __syncthreads();
    }
}

// ---------------- host orchestration ----------------
extern "C" void kernel_launch(void* const* d_in, const int* in_sizes, int n_in,
                              void* d_out, int out_size, void* d_ws, size_t ws_size,
                              hipStream_t stream) {
    const float* x        = (const float*)d_in[0];
    const float* emb_w    = (const float*)d_in[1];
    const float* emb_b    = (const float*)d_in[2];
    const float* in_w     = (const float*)d_in[3];
    const float* conv_w   = (const float*)d_in[4];
    const float* conv_b   = (const float*)d_in[5];
    const float* xp_w     = (const float*)d_in[6];
    const float* dtp_w    = (const float*)d_in[7];
    const float* dtp_b    = (const float*)d_in[8];
    const float* A_log    = (const float*)d_in[9];
    const float* Dparam   = (const float*)d_in[10];
    const float* out_w    = (const float*)d_in[11];
    const float* ln_w     = (const float*)d_in[12];
    const float* ln_b     = (const float*)d_in[13];
    const float* normf_w  = (const float*)d_in[14];
    const float* normf_b  = (const float*)d_in[15];
    const float* head_w   = (const float*)d_in[16];
    float* out = (float*)d_out;

    char* ws = (char*)d_ws;
    size_t off = 0;
    auto alloc = [&](size_t bytes) -> void* {
        off = (off + 255) & ~(size_t)255;
        void* p = ws + off;
        off += bytes;
        return p;
    };

    const size_t M = MTOK;
    // bf16 weights (padded where needed)
    unsigned short* wInBf   = (unsigned short*)alloc((size_t)NLAY * 2 * DI * DM * 2);
    unsigned short* wXpBf   = (unsigned short*)alloc((size_t)NLAY * 128 * DI * 2);   // 96 -> 128
    unsigned short* wDtBf   = (unsigned short*)alloc((size_t)NLAY * DI * DTR * 2);
    unsigned short* wOutBf  = (unsigned short*)alloc((size_t)NLAY * DM * DI * 2);
    unsigned short* wHeadBf = (unsigned short*)alloc((size_t)256 * DM * 2);          // 200 -> 256
    // activations (reused across layers)
    float*          hbuf  = (float*)alloc(M * DM * 4);
    float*          res   = (float*)alloc(M * DM * 4);
    unsigned short* hnBf  = (unsigned short*)alloc(M * DM * 2);
    float*          xz    = (float*)alloc(M * 2 * DI * 4);
    float*          xc    = (float*)alloc(M * DI * 4);
    unsigned short* xcBf  = (unsigned short*)alloc(M * DI * 2);
    float*          dbl   = (float*)alloc(M * 96 * 4);
    unsigned short* dtrBf = (unsigned short*)alloc(M * DTR * 2);
    float*          dt    = (float*)alloc(M * DI * 4);
    unsigned short* yBf   = (unsigned short*)alloc(M * DI * 2);
    unsigned short* finBf = (unsigned short*)alloc(M * DM * 2);
    (void)ws_size; (void)in_sizes; (void)n_in; (void)out_size;

    const int TPB = 256;
    auto blocks = [](size_t n, int t) { return (unsigned)((n + t - 1) / t); };

    // --- weights -> bf16 ---
    {
        size_t n = (size_t)NLAY * 2 * DI * DM;
        cvt_bf16_kernel<<<blocks(n, TPB), TPB, 0, stream>>>(in_w, wInBf, (int)n);
        n = (size_t)NLAY * DI * DTR;
        cvt_bf16_kernel<<<blocks(n, TPB), TPB, 0, stream>>>(dtp_w, wDtBf, (int)n);
        n = (size_t)NLAY * DM * DI;
        cvt_bf16_kernel<<<blocks(n, TPB), TPB, 0, stream>>>(out_w, wOutBf, (int)n);
        n = (size_t)NLAY * 128 * DI;
        cvt_pad_kernel<<<blocks(n, TPB), TPB, 0, stream>>>(xp_w, wXpBf, 96, 128, DI, (int)n);
        n = (size_t)256 * DM;
        cvt_pad_kernel<<<blocks(n, TPB), TPB, 0, stream>>>(head_w, wHeadBf, CITY, 256, DM, (int)n);
    }

    // --- embedding ---
    embed_kernel<<<blocks(M * DM, TPB), TPB, 0, stream>>>(x, emb_w, emb_b, hbuf, (int)(M * DM));

    const unsigned gy = (unsigned)(M / 256);  // 50

    for (int i = 0; i < NLAY; ++i) {
        // residual update + layernorm -> bf16
        residual_ln_kernel<<<(unsigned)M, 256, 0, stream>>>(
            hbuf, res, ln_w + (size_t)i * DM, ln_b + (size_t)i * DM, hnBf, i == 0);

        // xz = hn @ in_proj^T   (M x 4096, K=1024)
        gemm_bf16_wmma<<<dim3(2 * DI / 128, gy), 256, 0, stream>>>(
            (const __bf16*)hnBf, DM,
            (const __bf16*)(wInBf + (size_t)i * 2 * DI * DM), DM,
            xz, 2 * DI, DM, 2 * DI);

        // conv + silu -> xc (f32 + bf16)
        conv_silu_kernel<<<blocks(M * DI, TPB), TPB, 0, stream>>>(
            xz, conv_w + (size_t)i * DI * DCV, conv_b + (size_t)i * DI, xc, xcBf, (int)(M * DI));

        // dbl = xc @ x_proj^T   (M x 96, K=2048; weights padded to 128 rows)
        gemm_bf16_wmma<<<dim3(128 / 128, gy), 256, 0, stream>>>(
            (const __bf16*)xcBf, DI,
            (const __bf16*)(wXpBf + (size_t)i * 128 * DI), DI,
            dbl, 96, DI, 96);

        // dt_raw -> bf16
        dtraw_cvt_kernel<<<blocks(M * DTR, TPB), TPB, 0, stream>>>(dbl, dtrBf, (int)(M * DTR));

        // dt_lin = dt_raw @ dt_proj^T   (M x 2048, K=64)
        gemm_bf16_wmma<<<dim3(DI / 128, gy), 256, 0, stream>>>(
            (const __bf16*)dtrBf, DTR,
            (const __bf16*)(wDtBf + (size_t)i * DI * DTR), DTR,
            dt, DI, DTR, DI);

        // dt = softplus(dt_lin + b)
        dt_softplus_kernel<<<blocks(M * DI, TPB), TPB, 0, stream>>>(
            dt, dtp_b + (size_t)i * DI, (int)(M * DI));

        // selective scan + (y + u*D) * silu(z) -> bf16
        scan_kernel<<<dim3(DI / 256, BB), 256, 0, stream>>>(
            dbl, dt, xc, xz,
            A_log + (size_t)i * DI * NST, Dparam + (size_t)i * DI, yBf);

        // h = y @ out_proj^T   (M x 1024, K=2048)
        gemm_bf16_wmma<<<dim3(DM / 128, gy), 256, 0, stream>>>(
            (const __bf16*)yBf, DI,
            (const __bf16*)(wOutBf + (size_t)i * DM * DI), DI,
            hbuf, DM, DI, DM);
    }

    // final = h + residual; rmsnorm -> bf16
    final_rms_kernel<<<(unsigned)M, 256, 0, stream>>>(hbuf, res, normf_w, normf_b, finBf);

    // logits = final @ head^T   (M x 200, K=1024; weights padded to 256 rows)
    gemm_bf16_wmma<<<dim3(256 / 128, gy), 256, 0, stream>>>(
        (const __bf16*)finBf, DM,
        (const __bf16*)wHeadBf, DM,
        out, CITY, DM, CITY);
}
